// RNN_FeatureAttention_wc_37340445671486
// MI455X (gfx1250) — matrix operand order repaired and checked by reading, not compiled, not run on hardware
//
#include <hip/hip_runtime.h>
#include <hip/hip_bf16.h>
#include <math.h>

typedef _Float16 v16h __attribute__((ext_vector_type(16)));
typedef _Float16 v8h  __attribute__((ext_vector_type(8)));
typedef float    v8f  __attribute__((ext_vector_type(8)));

#define NB 256
#define NT 64
#define NH 24
#define DM 128
#define NHEADS 4
#define HD 32
#define FF 512

// ---- shared memory layout (bytes) ----
#define OFF_TOKF32 0           // 32x128 f32  = 16384
#define OFF_SCORES 16384       // 4x32x32 f32 = 16384
#define OFF_TOKF16 32768       // 32x128 f16  = 8192
#define OFF_QF16   40960       // 32x128 f16
#define OFF_KT16   49152       // 4x32(tok)x32(d) f16
#define OFF_VF16   57344       // 32x128 f16
#define OFF_ATTN16 65536       // 4x32x32 f16
#define OFF_AO16   73728       // 32x128 f16
#define OFF_FF16   81920       // 32x512 f16  = 32768
#define OFF_CONST  114688
// const area float offsets
#define CF_FE   0
#define CF_IPB  3072
#define CF_AOB  3456
#define CF_FB1  3584
#define CF_FB2  4096
#define CF_L1W  4224
#define CF_L1B  4352
#define CF_L2W  4480
#define CF_L2B  4608
#define CF_OW   4736
#define CF_WM   4864
#define CF_AM   5440
#define CF_BV   6016
#define CF_CV   6048
#define CF_OB   6080
#define CF_ZB   6088
#define CF_ZN   6120
#define CF_END  6152
#define SMEM_BYTES (OFF_CONST + CF_END * 4)   // 139296

// ---- workspace layout (bytes) ----
#define WS_PQKV 0          // 384x128 f16  = 98304 B
#define WS_PAO  98304      // 128x128 f16  = 32768 B
#define WS_PW1  131072     // 512x128 f16  = 131072 B
#define WS_PW2  262144     // 128x512 f16  = 131072 B
#define WS_XU   393216     // 256*64*24 f32 = 1572864 B

__device__ __forceinline__ v8f wmma16(v16h a, v16h b, v8f c) {
    return __builtin_amdgcn_wmma_f32_16x16x32_f16(false, a, false, b, (short)0, c,
                                                  false, false);
}

// A fragment (16x32 f16), rows m0..m0+15, K block k0..k0+31, from row-major LDS.
// ISA layout: lanes 0-15 hold row M=lane, K {0..7,16..23}; lanes 16-31 hold K {8..15,24..31}.
__device__ __forceinline__ v16h load_a(const _Float16* base, int stride, int m0,
                                       int k0, int lane) {
    const _Float16* p = base + (m0 + (lane & 15)) * stride + k0 + ((lane >> 4) << 3);
    v8h lo = *(const v8h*)p;
    v8h hi = *(const v8h*)(p + 16);
    v16h r;
#pragma unroll
    for (int i = 0; i < 8; ++i) { r[i] = lo[i]; r[8 + i] = hi[i]; }
    return r;
}

// -------- kernel 0: repack weights to f16 B-fragment order --------
// pack[o][k] = (f16) W[o][k]   (W is out x in, row-major)
__global__ void repack_kernel(const float* ipw, const float* aow, const float* w1,
                              const float* w2, _Float16* pQKV, _Float16* pAO,
                              _Float16* pW1, _Float16* pW2) {
    int idx = blockIdx.x * blockDim.x + threadIdx.x;
    const int n0 = 384 * 128, n1 = 128 * 128, n2 = 512 * 128, n3 = 128 * 512;
    if (idx < n0) {
        int o = idx / 128, i = idx % 128;
        pQKV[o * 128 + i] = (_Float16)ipw[idx];
    } else if (idx < n0 + n1) {
        int j = idx - n0; int o = j / 128, i = j % 128;
        pAO[o * 128 + i] = (_Float16)aow[j];
    } else if (idx < n0 + n1 + n2) {
        int j = idx - n0 - n1; int o = j / 128, i = j % 128;
        pW1[o * 128 + i] = (_Float16)w1[j];
    } else if (idx < n0 + n1 + n2 + n3) {
        int j = idx - n0 - n1 - n2; int o = j / 512, i = j % 512;
        pW2[o * 512 + i] = (_Float16)w2[j];
    }
}

// -------- kernel 1: xU[b,t,h] = x_in(b,t) . U[h,:]  (in_dim = 1 + 4) --------
__global__ void xu_kernel(const float* x_cont, const int* x_cat, const float* emb,
                          const float* U, float* xU) {
    int idx = blockIdx.x * blockDim.x + threadIdx.x;
    if (idx >= NB * NT * NH) return;
    int h = idx % NH;
    int bt = idx / NH;
    int cat = x_cat[bt];
    float acc = x_cont[bt] * U[h * 5 + 0];
#pragma unroll
    for (int j = 0; j < 4; ++j) acc += emb[cat * 4 + j] * U[h * 5 + 1 + j];
    xU[idx] = acc;
}

// -------- kernel 2: fused recurrent transformer block --------
__global__ __launch_bounds__(256) void rnn_attn_kernel(
    const float* __restrict__ xU, const float* __restrict__ Wm,
    const float* __restrict__ bvec, const float* __restrict__ Amat,
    const float* __restrict__ cvec, const float* __restrict__ fe_g,
    const float* __restrict__ ipb_g, const float* __restrict__ aob_g,
    const float* __restrict__ l1w_g, const float* __restrict__ l1b_g,
    const float* __restrict__ fb1_g, const float* __restrict__ fb2_g,
    const float* __restrict__ l2w_g, const float* __restrict__ l2b_g,
    const float* __restrict__ ow_g, const float* __restrict__ ob_g,
    const float* __restrict__ z0_g, const _Float16* __restrict__ pQKV,
    const _Float16* __restrict__ pAO, const _Float16* __restrict__ pW1,
    const _Float16* __restrict__ pW2, float* __restrict__ Oout,
    float* __restrict__ Zout) {
    extern __shared__ char smem[];
    float*    tokf32  = (float*)(smem + OFF_TOKF32);
    float*    scoresf = (float*)(smem + OFF_SCORES);
    _Float16* tokf16  = (_Float16*)(smem + OFF_TOKF16);
    _Float16* qf16    = (_Float16*)(smem + OFF_QF16);
    _Float16* kt16    = (_Float16*)(smem + OFF_KT16);
    _Float16* vf16    = (_Float16*)(smem + OFF_VF16);
    _Float16* at16    = (_Float16*)(smem + OFF_ATTN16);
    _Float16* ao16    = (_Float16*)(smem + OFF_AO16);
    _Float16* ff16    = (_Float16*)(smem + OFF_FF16);
    float*    cf      = (float*)(smem + OFF_CONST);

    const int tid  = threadIdx.x;
    const int lane = tid & 31;
    const int wav  = tid >> 5;
    const int blk  = blockIdx.x;

    // zero all LDS once (pads stay zero forever)
    for (int i = tid; i < SMEM_BYTES / 4; i += 256) ((unsigned int*)smem)[i] = 0u;
    __syncthreads();

    // stage constants into LDS
    for (int i = tid; i < NH * DM; i += 256) cf[CF_FE + i] = fe_g[i];
    for (int i = tid; i < 384; i += 256) cf[CF_IPB + i] = ipb_g[i];
    for (int i = tid; i < 512; i += 256) cf[CF_FB1 + i] = fb1_g[i];
    for (int i = tid; i < 576; i += 256) { cf[CF_WM + i] = Wm[i]; cf[CF_AM + i] = Amat[i]; }
    if (tid < 128) {
        cf[CF_AOB + tid] = aob_g[tid]; cf[CF_FB2 + tid] = fb2_g[tid];
        cf[CF_L1W + tid] = l1w_g[tid]; cf[CF_L1B + tid] = l1b_g[tid];
        cf[CF_L2W + tid] = l2w_g[tid]; cf[CF_L2B + tid] = l2b_g[tid];
        cf[CF_OW + tid]  = ow_g[tid];
    }
    if (tid < 24) { cf[CF_BV + tid] = bvec[tid]; cf[CF_CV + tid] = cvec[tid];
                    cf[CF_ZB + tid] = z0_g[tid]; }
    if (tid == 0) cf[CF_OB] = ob_g[0];
    __syncthreads();

    for (int t = 0; t < NT; ++t) {
        // warm L2 for packed weights (gfx1250 global_prefetch_b8)
        if (lane == 0) {
            __builtin_prefetch(pQKV + wav * 4096, 0, 1);
            __builtin_prefetch(pW1 + wav * 8192, 0, 1);
            __builtin_prefetch(pW2 + wav * 8192, 0, 1);
        }

        // ---- 1. recurrence: z = tanh(W @ z + xU_t + b) ----
        if (tid < NH) {
            float acc = cf[CF_BV + tid] + xU[((size_t)blk * NT + t) * NH + tid];
#pragma unroll 8
            for (int j = 0; j < NH; ++j) acc += cf[CF_WM + tid * NH + j] * cf[CF_ZB + j];
            cf[CF_ZN + tid] = tanhf(acc);
        }
        __syncthreads();
        if (tid < NH) cf[CF_ZB + tid] = cf[CF_ZN + tid];
        __syncthreads();

        // ---- 2. tokens = z[:,None] * feat_embed ----
        for (int i = tid; i < NH * DM; i += 256) {
            int m = i >> 7;
            float v = cf[CF_ZB + m] * cf[CF_FE + i];
            tokf32[i] = v;
            tokf16[i] = (_Float16)v;
        }
        __syncthreads();

        // ---- 3. QKV = tok @ in_proj_w.T + b  (32x384, 48 tiles x 4 ksteps) ----
        for (int tile = wav; tile < 48; tile += 8) {
            int mt = tile & 1, nt = tile >> 1;
            v8f acc = {};
#pragma unroll
            for (int kk = 0; kk < 4; ++kk) {
                v16h a = load_a(tokf16, DM, mt * 16, kk * 32, lane);
                v16h b = load_a(pQKV, DM, nt * 16, kk * 32, lane);
                acc = wmma16(a, b, acc);
            }
            int col = nt * 16 + (lane & 15);
            int rb  = (lane >> 4) * 8;
            int m0  = mt * 16 + rb;
            float bias = cf[CF_IPB + col];
            // nt is wave-uniform -> uniform branch over Q / K / V destinations
            if (nt < 8) {                       // Q: row-major f16
#pragma unroll
                for (int r = 0; r < 8; ++r)
                    qf16[(m0 + r) * DM + col] = (_Float16)(acc[r] + bias);
            } else if (nt < 16) {               // K: row-major [tok][d] per head
                int d = col - DM;
                _Float16* kcol = kt16 + ((d >> 5) * 32 + m0) * HD + (d & 31);
#pragma unroll
                for (int r = 0; r < 8; ++r)
                    kcol[r * HD] = (_Float16)(acc[r] + bias);
            } else {                            // V: row-major f16
                int d = col - 2 * DM;
#pragma unroll
                for (int r = 0; r < 8; ++r)
                    vf16[(m0 + r) * DM + d] = (_Float16)(acc[r] + bias);
            }
        }
        __syncthreads();

        // ---- 4. scores = q @ k^T * scale  (per head 32x32, 16 tiles) ----
        for (int tt = wav * 2; tt < wav * 2 + 2; ++tt) {
            int h = tt >> 2, rem = tt & 3, mt = rem >> 1, nl = rem & 1;
            v16h a = load_a(qf16, DM, mt * 16, h * HD, lane);
            v16h b = load_a(kt16 + (size_t)h * 32 * HD, HD, nl * 16, 0, lane);
            v8f acc = {};
            acc = wmma16(a, b, acc);
            int col = nl * 16 + (lane & 15);
            int rb  = (lane >> 4) * 8;
#pragma unroll
            for (int r = 0; r < 8; ++r)
                scoresf[((size_t)h * 32 + mt * 16 + rb + r) * 32 + col] =
                    acc[r] * 0.17677669529663687f;  // 1/sqrt(32)
        }
        __syncthreads();

        // ---- 5. softmax over 24 keys -> f16 A-fragment matrix ----
        if (tid < NHEADS * NH) {
            int h = tid / NH, m = tid % NH;
            float* srow = scoresf + ((size_t)h * 32 + m) * 32;
            float mx = srow[0];
#pragma unroll 8
            for (int n = 1; n < NH; ++n) mx = fmaxf(mx, srow[n]);
            float s = 0.f;
#pragma unroll 8
            for (int n = 0; n < NH; ++n) { float e = __expf(srow[n] - mx); srow[n] = e; s += e; }
            float inv = 1.f / s;
            _Float16* arow = at16 + ((size_t)h * 32 + m) * 32;
#pragma unroll 8
            for (int n = 0; n < NH; ++n) arow[n] = (_Float16)(srow[n] * inv);
#pragma unroll
            for (int n = NH; n < 32; ++n) arow[n] = (_Float16)0.f;
        }
        __syncthreads();

        // ---- 6. ao = attn @ v  (per head 32x32, 16 tiles) ----
        for (int tt = wav * 2; tt < wav * 2 + 2; ++tt) {
            int h = tt >> 2, rem = tt & 3, mt = rem >> 1, nl = rem & 1;
            v16h a = load_a(at16 + (size_t)h * HD * 32, 32, mt * 16, 0, lane);
            v16h b;
#pragma unroll
            for (int i = 0; i < 8; ++i) {
                b[i] = vf16[(size_t)(((lane >> 4) << 3) + i) * DM + h * HD + nl * 16 + (lane & 15)];
                b[8 + i] = vf16[(size_t)(16 + ((lane >> 4) << 3) + i) * DM + h * HD + nl * 16 + (lane & 15)];
            }
            v8f acc = {};
            acc = wmma16(a, b, acc);
            int col = h * HD + nl * 16 + (lane & 15);
            int rb  = (lane >> 4) * 8;
#pragma unroll
            for (int r = 0; r < 8; ++r)
                ao16[(size_t)(mt * 16 + rb + r) * DM + col] = (_Float16)acc[r];
        }
        __syncthreads();

        // ---- 7. attn-out proj + residual into tokf32 (16 tiles x 4 ksteps) ----
        for (int tile = wav; tile < 16; tile += 8) {
            int mt = tile & 1, nt = tile >> 1;
            v8f acc = {};
#pragma unroll
            for (int kk = 0; kk < 4; ++kk) {
                v16h a = load_a(ao16, DM, mt * 16, kk * 32, lane);
                v16h b = load_a(pAO, DM, nt * 16, kk * 32, lane);
                acc = wmma16(a, b, acc);
            }
            int col = nt * 16 + (lane & 15);
            int rb  = (lane >> 4) * 8;
            float bias = cf[CF_AOB + col];
#pragma unroll
            for (int r = 0; r < 8; ++r) {
                int m = mt * 16 + rb + r;
                tokf32[m * DM + col] = acc[r] + bias + tokf32[m * DM + col];
            }
        }
        __syncthreads();

        // ---- 8. LayerNorm 1 (rows < 24), write f32 + f16 ----
        if (tid < NH) {
            float* row = tokf32 + tid * DM;
            float mean = 0.f;
#pragma unroll 8
            for (int d = 0; d < DM; ++d) mean += row[d];
            mean *= (1.f / DM);
            float var = 0.f;
#pragma unroll 8
            for (int d = 0; d < DM; ++d) { float df = row[d] - mean; var += df * df; }
            float inv = rsqrtf(var * (1.f / DM) + 1e-5f);
#pragma unroll 8
            for (int d = 0; d < DM; ++d) {
                float y = (row[d] - mean) * inv * cf[CF_L1W + d] + cf[CF_L1B + d];
                row[d] = y;
                tokf16[tid * DM + d] = (_Float16)y;
            }
        }
        __syncthreads();

        // ---- 9. FFN1 + ReLU -> ff16 (64 tiles x 4 ksteps) ----
        for (int tile = wav; tile < 64; tile += 8) {
            int mt = tile & 1, nt = tile >> 1;
            v8f acc = {};
#pragma unroll
            for (int kk = 0; kk < 4; ++kk) {
                v16h a = load_a(tokf16, DM, mt * 16, kk * 32, lane);
                v16h b = load_a(pW1, DM, nt * 16, kk * 32, lane);
                acc = wmma16(a, b, acc);
            }
            int col = nt * 16 + (lane & 15);
            int rb  = (lane >> 4) * 8;
            float bias = cf[CF_FB1 + col];
#pragma unroll
            for (int r = 0; r < 8; ++r) {
                int m = mt * 16 + rb + r;
                ff16[(size_t)m * FF + col] = (_Float16)fmaxf(acc[r] + bias, 0.f);
            }
        }
        __syncthreads();

        // ---- 10. FFN2 + residual (16 tiles x 16 ksteps) ----
        for (int tile = wav; tile < 16; tile += 8) {
            int mt = tile & 1, nt = tile >> 1;
            v8f acc = {};
#pragma unroll
            for (int kk = 0; kk < 16; ++kk) {
                v16h a = load_a(ff16, FF, mt * 16, kk * 32, lane);
                v16h b = load_a(pW2, FF, nt * 16, kk * 32, lane);
                acc = wmma16(a, b, acc);
            }
            int col = nt * 16 + (lane & 15);
            int rb  = (lane >> 4) * 8;
            float bias = cf[CF_FB2 + col];
#pragma unroll
            for (int r = 0; r < 8; ++r) {
                int m = mt * 16 + rb + r;
                tokf32[m * DM + col] = acc[r] + bias + tokf32[m * DM + col];
            }
        }
        __syncthreads();

        // ---- 11. LayerNorm 2 ----
        if (tid < NH) {
            float* row = tokf32 + tid * DM;
            float mean = 0.f;
#pragma unroll 8
            for (int d = 0; d < DM; ++d) mean += row[d];
            mean *= (1.f / DM);
            float var = 0.f;
#pragma unroll 8
            for (int d = 0; d < DM; ++d) { float df = row[d] - mean; var += df * df; }
            float inv = rsqrtf(var * (1.f / DM) + 1e-5f);
#pragma unroll 8
            for (int d = 0; d < DM; ++d)
                row[d] = (row[d] - mean) * inv * cf[CF_L2W + d] + cf[CF_L2B + d];
        }
        __syncthreads();

        // ---- 12. delta, z update, outputs ----
        if (tid < NH) {
            const float* row = tokf32 + tid * DM;
            float acc = cf[CF_OB];
#pragma unroll 8
            for (int d = 0; d < DM; ++d) acc += row[d] * cf[CF_OW + d];
            cf[CF_ZN + tid] = cf[CF_ZB + tid] + acc;
        }
        __syncthreads();
        if (tid < NH) {
            float zi = cf[CF_ZN + tid];
            float o = cf[CF_CV + tid];
#pragma unroll 8
            for (int j = 0; j < NH; ++j) o += cf[CF_AM + tid * NH + j] * cf[CF_ZN + j];
            size_t base = ((size_t)blk * NT + t) * NH + tid;
            Oout[base] = o;
            Zout[base] = zi;
            cf[CF_ZB + tid] = zi;
        }
        __syncthreads();
    }
}

extern "C" void kernel_launch(void* const* d_in, const int* in_sizes, int n_in,
                              void* d_out, int out_size, void* d_ws, size_t ws_size,
                              hipStream_t stream) {
    const float* x_cont = (const float*)d_in[0];
    const int*   x_cat  = (const int*)d_in[1];
    const float* emb    = (const float*)d_in[2];
    const float* Wm     = (const float*)d_in[3];
    const float* U      = (const float*)d_in[4];
    const float* bvec   = (const float*)d_in[5];
    const float* Amat   = (const float*)d_in[6];
    const float* cvec   = (const float*)d_in[7];
    const float* fe     = (const float*)d_in[8];
    const float* ipw    = (const float*)d_in[9];
    const float* ipb    = (const float*)d_in[10];
    const float* aow    = (const float*)d_in[11];
    const float* aob    = (const float*)d_in[12];
    const float* l1w    = (const float*)d_in[13];
    const float* l1b    = (const float*)d_in[14];
    const float* w1     = (const float*)d_in[15];
    const float* fb1    = (const float*)d_in[16];
    const float* w2     = (const float*)d_in[17];
    const float* fb2    = (const float*)d_in[18];
    const float* l2w    = (const float*)d_in[19];
    const float* l2b    = (const float*)d_in[20];
    const float* ow     = (const float*)d_in[21];
    const float* ob     = (const float*)d_in[22];
    const float* z0     = (const float*)d_in[23];

    char* ws = (char*)d_ws;
    _Float16* pQKV = (_Float16*)(ws + WS_PQKV);
    _Float16* pAO  = (_Float16*)(ws + WS_PAO);
    _Float16* pW1  = (_Float16*)(ws + WS_PW1);
    _Float16* pW2  = (_Float16*)(ws + WS_PW2);
    float*    xU   = (float*)(ws + WS_XU);

    repack_kernel<<<(384 * 128 + 128 * 128 + 512 * 128 + 128 * 512 + 255) / 256, 256, 0,
                    stream>>>(ipw, aow, w1, w2, pQKV, pAO, pW1, pW2);
    xu_kernel<<<(NB * NT * NH + 255) / 256, 256, 0, stream>>>(x_cont, x_cat, emb, U, xU);

    float* O = (float*)d_out;
    float* Z = O + (size_t)NB * NT * NH;
    rnn_attn_kernel<<<NB, 256, SMEM_BYTES, stream>>>(
        xU, Wm, bvec, Amat, cvec, fe, ipb, aob, l1w, l1b, fb1, fb2, l2w, l2b, ow, ob,
        z0, pQKV, pAO, pW1, pW2, O, Z);
}
